// PerformerSelfAttention_26671746908378
// MI455X (gfx1250) — compile-verified, matrix operand order checked by reading, not run on hardware
//
#include <hip/hip_runtime.h>
#include <stdint.h>

// ---------------------------------------------------------------------------
// Problem constants (match reference)
// ---------------------------------------------------------------------------
#define BB 4
#define TT 2048
#define DD 1024
#define HH 16
#define DH 64
#define ROWS_BT (BB * TT)          // 8192
#define ROWS_BTH (BB * TT * HH)    // 131072

typedef unsigned short u16;
typedef __attribute__((ext_vector_type(16))) __bf16 v16bf;
typedef __attribute__((ext_vector_type(8)))  float  v8f;

union FragBF { v16bf v; uint4 q[2]; };

__device__ __forceinline__ u16 f2bf(float f) {
  union { float f; uint32_t u; } a; a.f = f;
  uint32_t u = a.u;
  u += 0x7FFFu + ((u >> 16) & 1u);   // round-to-nearest-even
  return (u16)(u >> 16);
}

__device__ __forceinline__ float wave_sum(float v) {
  #pragma unroll
  for (int m = 16; m; m >>= 1) v += __shfl_xor(v, m, 32);
  return v;
}

__device__ __forceinline__ float block_sum256(float v, float* sm) {
  v = wave_sum(v);
  int wave = threadIdx.x >> 5, lane = threadIdx.x & 31;
  __syncthreads();
  if (lane == 0) sm[wave] = v;
  __syncthreads();
  float t = 0.f;
  #pragma unroll
  for (int i = 0; i < 8; ++i) t += sm[i];
  return t;
}

// ---------------------------------------------------------------------------
// fp32 -> bf16 convert (grid-stride, row-major preserved) -- activations
// ---------------------------------------------------------------------------
__global__ __launch_bounds__(256) void k_f32_to_bf16(const float* __restrict__ in,
                                                     u16* __restrict__ out, long n) {
  long i = (long)blockIdx.x * blockDim.x + threadIdx.x;
  long stride = (long)gridDim.x * blockDim.x;
  for (; i < n; i += stride) out[i] = f2bf(in[i]);
}

__global__ __launch_bounds__(256) void k_silu_to_bf16(const float* __restrict__ in,
                                                      u16* __restrict__ out, long n) {
  long i = (long)blockIdx.x * blockDim.x + threadIdx.x;
  long stride = (long)gridDim.x * blockDim.x;
  for (; i < n; i += stride) {
    float x = in[i];
    out[i] = f2bf(x / (1.f + expf(-x)));
  }
}

// ---------------------------------------------------------------------------
// Weight convert + transpose:  W[K][N] fp32  ->  out[N][K] bf16
//   32x32 LDS tiles, coalesced on both sides.  grid = (N/32, K/32)
// ---------------------------------------------------------------------------
__global__ __launch_bounds__(256) void k_w_to_bf16T(const float* __restrict__ W,
                                                    u16* __restrict__ out,
                                                    int K, int N) {
  __shared__ u16 sT[32 * 33];
  int nt = blockIdx.x, kt = blockIdx.y;
  int rl = threadIdx.x >> 5;   // 0..7
  int cl = threadIdx.x & 31;   // 0..31
  #pragma unroll
  for (int r = 0; r < 4; ++r) {
    int krow = kt * 32 + rl + r * 8;
    sT[cl * 33 + rl + r * 8] = f2bf(W[(size_t)krow * N + nt * 32 + cl]);
  }
  __syncthreads();
  #pragma unroll
  for (int r = 0; r < 4; ++r) {
    int nl = rl + r * 8;
    out[(size_t)(nt * 32 + nl) * K + kt * 32 + cl] = sT[nl * 33 + cl];
  }
}

// ---------------------------------------------------------------------------
// Pre-LN over D=1024, emit bf16
// ---------------------------------------------------------------------------
__global__ __launch_bounds__(256) void k_ln_pre(const float* __restrict__ x,
                                                const float* __restrict__ g,
                                                const float* __restrict__ b,
                                                u16* __restrict__ out) {
  __shared__ float sm[8];
  long row = blockIdx.x;
  const float* xp = x + row * DD;
  int c = threadIdx.x * 4;
  float4 v = *(const float4*)(xp + c);
  float mean = block_sum256(v.x + v.y + v.z + v.w, sm) * (1.f / DD);
  float dx = v.x - mean, dy = v.y - mean, dz = v.z - mean, dw = v.w - mean;
  float var = block_sum256(dx*dx + dy*dy + dz*dz + dw*dw, sm) * (1.f / DD);
  float rstd = rsqrtf(var + 1e-5f);
  float y0 = dx * rstd * g[c+0] + b[c+0];
  float y1 = dy * rstd * g[c+1] + b[c+1];
  float y2 = dz * rstd * g[c+2] + b[c+2];
  float y3 = dw * rstd * g[c+3] + b[c+3];
  uint2 p;
  p.x = (uint32_t)f2bf(y0) | ((uint32_t)f2bf(y1) << 16);
  p.y = (uint32_t)f2bf(y2) | ((uint32_t)f2bf(y3) << 16);
  *(uint2*)(out + row * DD + c) = p;
}

// ---------------------------------------------------------------------------
// WMMA bf16 GEMM:  C[M,N] = epilogue( (A[M,K] @ B[K,N] + bias) * alpha )
//   A: [M][K] bf16 row-major;  BT: [N][K] bf16 (pre-transposed weight)
//   block tile 128x64, 8 waves, wave = 32x32 via 4x v_wmma_f32_16x16x32_bf16
//   Double-buffered LDS; global_prefetch for tile-after-next.
//   mode 0: none | 1: exact GELU | 2: C = res + 0.1*val (residual)
// ---------------------------------------------------------------------------
#define GTM 128
#define GTN 64
#define GTK 32
#define LDA_S 40
#define LDB_S 40

__global__ __launch_bounds__(256) void k_gemm_bf16(
    const u16* __restrict__ A, const u16* __restrict__ BT,
    const float* __restrict__ bias, float* __restrict__ C,
    int M, int N, int K, float alpha, int mode, const float* __restrict__ res) {
  __shared__ u16 sA[2][GTM * LDA_S];
  __shared__ u16 sB[2][GTN * LDB_S];
  int tid  = threadIdx.x;
  int lane = tid & 31;
  int wave = tid >> 5;
  int waveM = (wave >> 1) * 32;
  int waveN = (wave & 1) * 32;
  int m0 = blockIdx.y * GTM;
  int n0 = blockIdx.x * GTN;

  v8f acc[2][2];
  #pragma unroll
  for (int mt = 0; mt < 2; ++mt)
    #pragma unroll
    for (int nt = 0; nt < 2; ++nt)
      acc[mt][nt] = (v8f){0.f,0.f,0.f,0.f,0.f,0.f,0.f,0.f};

  int arow = tid >> 1;           // 0..127
  int aseg = (tid & 1) * 16;     // 0 or 16 (u16 units)
  int brow = tid >> 2;           // 0..63
  int bseg = (tid & 3) * 8;      // 0,8,16,24

  const u16* aP = A  + (size_t)(m0 + arow) * K + aseg;
  const u16* bP = BT + (size_t)(n0 + brow) * K + bseg;
  const bool aok = (m0 + arow) < M;

  uint4 ra0, ra1, rb;
  auto ldg = [&](int k0) {
    if (aok) {
      const uint4* gp = (const uint4*)(aP + k0);
      ra0 = gp[0]; ra1 = gp[1];
    } else {
      ra0 = make_uint4(0,0,0,0); ra1 = make_uint4(0,0,0,0);
    }
    rb = *(const uint4*)(bP + k0);
  };
  auto stl = [&](int bf) {
    *(uint4*)(&sA[bf][arow * LDA_S + aseg])     = ra0;
    *(uint4*)(&sA[bf][arow * LDA_S + aseg + 8]) = ra1;
    *(uint4*)(&sB[bf][brow * LDB_S + bseg])     = rb;
  };

  ldg(0);
  stl(0);
  __syncthreads();

  int buf = 0;
  for (int k0 = 0; k0 < K; k0 += GTK) {
    int kn = k0 + GTK;
    if (kn < K) ldg(kn);                       // overlap next-tile loads with math
    if (kn + GTK < K) {                        // warm GL2 two tiles ahead
      __builtin_prefetch(aP + kn + GTK, 0, 1);
      __builtin_prefetch(bP + kn + GTK, 0, 1);
    }

    int base = (lane >> 4) * 8;                // K half per lane group
    FragBF af[2], bfg[2];
    #pragma unroll
    for (int mt = 0; mt < 2; ++mt) {
      int r = waveM + mt * 16 + (lane & 15);
      af[mt].q[0] = *(const uint4*)(&sA[buf][r * LDA_S + base]);
      af[mt].q[1] = *(const uint4*)(&sA[buf][r * LDA_S + base + 16]);
    }
    #pragma unroll
    for (int nt = 0; nt < 2; ++nt) {
      int cidx = waveN + nt * 16 + (lane & 15);
      bfg[nt].q[0] = *(const uint4*)(&sB[buf][cidx * LDB_S + base]);
      bfg[nt].q[1] = *(const uint4*)(&sB[buf][cidx * LDB_S + base + 16]);
    }
    #pragma unroll
    for (int mt = 0; mt < 2; ++mt)
      #pragma unroll
      for (int nt = 0; nt < 2; ++nt)
        acc[mt][nt] = __builtin_amdgcn_wmma_f32_16x16x32_bf16(
            false, af[mt].v, false, bfg[nt].v, (short)0, acc[mt][nt], false, false);

    if (kn < K) stl(buf ^ 1);
    __syncthreads();
    buf ^= 1;
  }

  // --- epilogue ---
  int mlane = (lane >> 4) * 8;
  int nlane = lane & 15;
  #pragma unroll
  for (int mt = 0; mt < 2; ++mt)
    #pragma unroll
    for (int nt = 0; nt < 2; ++nt)
      #pragma unroll
      for (int v = 0; v < 8; ++v) {
        int row = m0 + waveM + mt * 16 + mlane + v;
        int col = n0 + waveN + nt * 16 + nlane;
        if (row < M) {
          float val = (acc[mt][nt][v] + bias[col]) * alpha;
          size_t idx = (size_t)row * N + col;
          if (mode == 1)      val = 0.5f * val * (1.f + erff(val * 0.70710678118f));
          else if (mode == 2) val = res[idx] + 0.1f * val;
          C[idx] = val;
        }
      }
}

// ---------------------------------------------------------------------------
// Performer per-head row kernel (wave per 64-elem row, in place)
//   mode 0: q (LN + l2norm + exp(clip(@P))*0.1)
//   mode 1: k (same + mask)
//   mode 2: v (LN only)
// ---------------------------------------------------------------------------
__global__ __launch_bounds__(256) void k_perf(float* __restrict__ buf,
                                              const float* __restrict__ P,
                                              const float* __restrict__ g,
                                              const float* __restrict__ b,
                                              const float* __restrict__ mask,
                                              int mode) {
  __shared__ float sP[DH * DH];
  for (int i = threadIdx.x; i < DH * DH; i += 256) sP[i] = P[i];
  __syncthreads();

  int lane = threadIdx.x & 31, wave = threadIdx.x >> 5;
  long row = (long)blockIdx.x * 8 + wave;   // < ROWS_BTH
  float* rp = buf + row * DH;
  float x0 = rp[lane], x1 = rp[lane + 32];
  float mean = wave_sum(x0 + x1) * (1.f / DH);
  float d0 = x0 - mean, d1 = x1 - mean;
  float var = wave_sum(d0*d0 + d1*d1) * (1.f / DH);
  float rstd = rsqrtf(var + 1e-5f);
  float y0 = d0 * rstd * g[lane]      + b[lane];
  float y1 = d1 * rstd * g[lane + 32] + b[lane + 32];

  if (mode == 2) { rp[lane] = y0; rp[lane + 32] = y1; return; }

  float n2  = wave_sum(y0*y0 + y1*y1);
  float inv = 1.f / fmaxf(sqrtf(n2), 1e-12f);
  y0 *= inv; y1 *= inv;

  float a0 = 0.f, a1 = 0.f;
  #pragma unroll
  for (int n = 0; n < DH; ++n) {
    float yv = __shfl(n < 32 ? y0 : y1, n & 31, 32);
    a0 = fmaf(yv, sP[n * DH + lane],      a0);
    a1 = fmaf(yv, sP[n * DH + lane + 32], a1);
  }
  float e0 = expf(fminf(fmaxf(a0, -15.f), 15.f)) * 0.1f;
  float e1 = expf(fminf(fmaxf(a1, -15.f), 15.f)) * 0.1f;
  if (mode == 1) {
    float mk = mask[row >> 4];   // row/H = b*T+t
    e0 *= mk; e1 *= mk;
  }
  rp[lane] = e0; rp[lane + 32] = e1;
}

// ---------------------------------------------------------------------------
// kv[b,h][m][n] = 0.1 * sum_t kp[b,t,h,m] * v[b,t,h,n]   (block per (b,h))
// ---------------------------------------------------------------------------
__global__ __launch_bounds__(256) void k_kv(const float* __restrict__ kp,
                                            const float* __restrict__ vn,
                                            float* __restrict__ kvout) {
  __shared__ float sK[16 * DH], sV[16 * DH];
  int bh = blockIdx.x;
  int b = bh >> 4, h = bh & 15;
  int tid = threadIdx.x;
  int m  = tid >> 2;
  int nb = (tid & 3) * 16;
  float acc[16];
  #pragma unroll
  for (int j = 0; j < 16; ++j) acc[j] = 0.f;
  long rowbase = ((long)b * TT) * (long)DD + (long)h * DH;
  for (int t0 = 0; t0 < TT; t0 += 16) {
    __syncthreads();
    for (int e = tid; e < 16 * DH; e += 256) {
      int i = e >> 6, j = e & 63;
      long off = rowbase + (long)(t0 + i) * DD + j;
      sK[e] = kp[off]; sV[e] = vn[off];
    }
    __syncthreads();
    #pragma unroll 4
    for (int i = 0; i < 16; ++i) {
      float km = sK[i * DH + m];
      #pragma unroll
      for (int j = 0; j < 16; ++j) acc[j] = fmaf(km, sV[i * DH + nb + j], acc[j]);
    }
  }
  float* o = kvout + (long)bh * (DH * DH) + m * DH + nb;
  #pragma unroll
  for (int j = 0; j < 16; ++j) o[j] = acc[j] * 0.1f;
}

// ---------------------------------------------------------------------------
// qkv + denom + LN, wave per token, writes (B,T,H,DH) == (B,T,D)
// ---------------------------------------------------------------------------
__global__ __launch_bounds__(256) void k_qkv(const float* __restrict__ qp,
                                             const float* __restrict__ kp,
                                             const float* __restrict__ kvb,
                                             const float* __restrict__ g,
                                             const float* __restrict__ b,
                                             float* __restrict__ attn) {
  __shared__ float sKV[DH * DH];
  int bh = blockIdx.y;
  for (int e = threadIdx.x; e < DH * DH; e += 256) sKV[e] = kvb[(long)bh * (DH * DH) + e];
  __syncthreads();

  int lane = threadIdx.x & 31, wave = threadIdx.x >> 5;
  int b_ = bh >> 4, h = bh & 15;
  int t = blockIdx.x * 8 + wave;
  long roff = (((long)b_ * TT + t) * HH + h) * DH;
  float q0 = qp[roff + lane], q1 = qp[roff + lane + 32];
  float k0 = kp[roff + lane], k1 = kp[roff + lane + 32];
  float dn = fmaxf(wave_sum(q0 * k0 + q1 * k1), 1e-6f);

  float a0 = 0.f, a1 = 0.f;
  #pragma unroll
  for (int n = 0; n < DH; ++n) {
    float qv = __shfl(n < 32 ? q0 : q1, n & 31, 32);
    a0 = fmaf(qv, sKV[n * DH + lane],      a0);
    a1 = fmaf(qv, sKV[n * DH + lane + 32], a1);
  }
  a0 = a0 * 0.1f / dn;
  a1 = a1 * 0.1f / dn;
  float mean = wave_sum(a0 + a1) * (1.f / DH);
  float d0 = a0 - mean, d1 = a1 - mean;
  float var = wave_sum(d0*d0 + d1*d1) * (1.f / DH);
  float rstd = rsqrtf(var + 1e-5f);
  attn[roff + lane]      = d0 * rstd * g[lane]      + b[lane];
  attn[roff + lane + 32] = d1 * rstd * g[lane + 32] + b[lane + 32];
}

// ---------------------------------------------------------------------------
// Fused: post-LN -> l2norm*sqrt(D) -> style-LN -> (1+scale)+shift -> SiLU -> bf16
// ---------------------------------------------------------------------------
__global__ __launch_bounds__(256) void k_post_styl(const float* __restrict__ a2,
                                                   const float* __restrict__ gp,
                                                   const float* __restrict__ bp,
                                                   const float* __restrict__ gs,
                                                   const float* __restrict__ bs,
                                                   const float* __restrict__ embout,
                                                   u16* __restrict__ hsbf) {
  __shared__ float sm[8];
  long row = blockIdx.x;
  long b = row / TT;
  int c = threadIdx.x * 4;
  float4 v = *(const float4*)(a2 + row * DD + c);
  float xv[4] = {v.x, v.y, v.z, v.w};

  float mean = block_sum256(xv[0]+xv[1]+xv[2]+xv[3], sm) * (1.f / DD);
  float acc = 0.f;
  #pragma unroll
  for (int i = 0; i < 4; ++i) { xv[i] -= mean; acc += xv[i]*xv[i]; }
  float rstd = rsqrtf(block_sum256(acc, sm) * (1.f / DD) + 1e-5f);
  float u[4];
  #pragma unroll
  for (int i = 0; i < 4; ++i) u[i] = xv[i] * rstd * gp[c+i] + bp[c+i];

  float ss = block_sum256(u[0]*u[0]+u[1]*u[1]+u[2]*u[2]+u[3]*u[3], sm);
  float l2s = 32.f / fmaxf(sqrtf(ss), 1e-12f);   // sqrt(D)=32
  #pragma unroll
  for (int i = 0; i < 4; ++i) u[i] *= l2s;

  float mean2 = block_sum256(u[0]+u[1]+u[2]+u[3], sm) * (1.f / DD);
  acc = 0.f;
  #pragma unroll
  for (int i = 0; i < 4; ++i) { u[i] -= mean2; acc += u[i]*u[i]; }
  float rstd2 = rsqrtf(block_sum256(acc, sm) * (1.f / DD) + 1e-5f);

  const float* eo = embout + b * (2 * DD);
  u16 s16[4];
  #pragma unroll
  for (int i = 0; i < 4; ++i) {
    float z = u[i] * rstd2 * gs[c+i] + bs[c+i];
    float hs = z * (1.f + eo[c+i]) + eo[DD + c + i];
    s16[i] = f2bf(hs / (1.f + expf(-hs)));
  }
  uint2 p;
  p.x = (uint32_t)s16[0] | ((uint32_t)s16[1] << 16);
  p.y = (uint32_t)s16[2] | ((uint32_t)s16[3] << 16);
  *(uint2*)(hsbf + row * DD + c) = p;
}

// ---------------------------------------------------------------------------
// Launch
// ---------------------------------------------------------------------------
extern "C" void kernel_launch(void* const* d_in, const int* in_sizes, int n_in,
                              void* d_out, int out_size, void* d_ws, size_t ws_size,
                              hipStream_t stream) {
  const float* x      = (const float*)d_in[0];
  const float* emb    = (const float*)d_in[1];
  const float* mask   = (const float*)d_in[2];
  const float* g_pre  = (const float*)d_in[3];
  const float* b_pre  = (const float*)d_in[4];
  const float* Wq     = (const float*)d_in[5];
  const float* bq     = (const float*)d_in[6];
  const float* Wk     = (const float*)d_in[7];
  const float* bk     = (const float*)d_in[8];
  const float* Wv     = (const float*)d_in[9];
  const float* bv     = (const float*)d_in[10];
  const float* P      = (const float*)d_in[11];
  const float* g_fa   = (const float*)d_in[12];
  const float* b_fa   = (const float*)d_in[13];
  const float* W1     = (const float*)d_in[14];
  const float* b1     = (const float*)d_in[15];
  const float* W2     = (const float*)d_in[16];
  const float* b2     = (const float*)d_in[17];
  const float* g_post = (const float*)d_in[18];
  const float* b_post = (const float*)d_in[19];
  const float* W_emb  = (const float*)d_in[20];
  const float* b_emb  = (const float*)d_in[21];
  const float* g_st   = (const float*)d_in[22];
  const float* b_st   = (const float*)d_in[23];
  const float* W_out  = (const float*)d_in[24];
  const float* b_out  = (const float*)d_in[25];
  float* out = (float*)d_out;

  char* w = (char*)d_ws;
  u16*   hbf    = (u16*)  (w + 0);                      // 16 MB  (8192x1024 bf16)
  u16*   wbf    = (u16*)  (w + (16ull  << 20));         //  4 MB  (weight staging, [N][K])
  float* qf     = (float*)(w + (20ull  << 20));         // 32 MB
  float* kf     = (float*)(w + (52ull  << 20));         // 32 MB
  float* vf     = (float*)(w + (84ull  << 20));         // 32 MB
  float* kvb    = (float*)(w + (116ull << 20));         //  1 MB
  float* attnf  = (float*)(w + (117ull << 20));         // 32 MB
  u16*   abf    = (u16*)  (w + (149ull << 20));         // 16 MB  (act staging)
  float* g1f    = (float*)(w + (165ull << 20));         // 32 MB
  u16*   embbf  = (u16*)  (w + (197ull << 20));         //  8 KB
  float* embout = (float*)(w + (198ull << 20));         // 32 KB

  const dim3 gGemm(DD / GTN, ROWS_BT / GTM);   // (16, 64)
  const dim3 gWT(DD / 32, DD / 32);            // (32, 32) for square weights

  // 1) pre-LN -> bf16
  k_ln_pre<<<ROWS_BT, 256, 0, stream>>>(x, g_pre, b_pre, hbf);

  // 2) QKV GEMMs (alpha = 0.1 applied to acc+bias)
  k_w_to_bf16T<<<gWT, 256, 0, stream>>>(Wq, wbf, DD, DD);
  k_gemm_bf16<<<gGemm, 256, 0, stream>>>(hbf, wbf, bq, qf, ROWS_BT, DD, DD, 0.1f, 0, nullptr);
  k_w_to_bf16T<<<gWT, 256, 0, stream>>>(Wk, wbf, DD, DD);
  k_gemm_bf16<<<gGemm, 256, 0, stream>>>(hbf, wbf, bk, kf, ROWS_BT, DD, DD, 0.1f, 0, nullptr);
  k_w_to_bf16T<<<gWT, 256, 0, stream>>>(Wv, wbf, DD, DD);
  k_gemm_bf16<<<gGemm, 256, 0, stream>>>(hbf, wbf, bv, vf, ROWS_BT, DD, DD, 0.1f, 0, nullptr);

  // 3) Performer feature maps (in place)
  k_perf<<<ROWS_BTH / 8, 256, 0, stream>>>(qf, P, g_fa, b_fa, mask, 0);
  k_perf<<<ROWS_BTH / 8, 256, 0, stream>>>(kf, P, g_fa, b_fa, mask, 1);
  k_perf<<<ROWS_BTH / 8, 256, 0, stream>>>(vf, P, g_fa, b_fa, mask, 2);

  // 4) kv and qkv (+denom, +LN) -> attn (B,T,D)
  k_kv<<<BB * HH, 256, 0, stream>>>(kf, vf, kvb);
  k_qkv<<<dim3(TT / 8, BB * HH), 256, 0, stream>>>(qf, kf, kvb, g_fa, b_fa, attnf);

  // 5) FFN: GELU(attn@W1+b1)@W2+b2
  k_f32_to_bf16<<<4096, 256, 0, stream>>>(attnf, abf, (long)ROWS_BT * DD);
  k_w_to_bf16T<<<gWT, 256, 0, stream>>>(W1, wbf, DD, DD);
  k_gemm_bf16<<<gGemm, 256, 0, stream>>>(abf, wbf, b1, g1f, ROWS_BT, DD, DD, 1.0f, 1, nullptr);
  k_f32_to_bf16<<<4096, 256, 0, stream>>>(g1f, abf, (long)ROWS_BT * DD);
  k_w_to_bf16T<<<gWT, 256, 0, stream>>>(W2, wbf, DD, DD);
  k_gemm_bf16<<<gGemm, 256, 0, stream>>>(abf, wbf, b2, attnf, ROWS_BT, DD, DD, 1.0f, 0, nullptr);

  // 6) stylization scale/shift: silu(emb) @ W_emb + b_emb  (M=4, N=2048)
  k_silu_to_bf16<<<16, 256, 0, stream>>>(emb, embbf, (long)BB * 1024);
  k_w_to_bf16T<<<dim3(2048 / 32, DD / 32), 256, 0, stream>>>(W_emb, wbf, DD, 2048);
  k_gemm_bf16<<<dim3(2048 / GTN, 1), 256, 0, stream>>>(embbf, wbf, b_emb, embout,
                                                       BB, 2048, DD, 1.0f, 0, nullptr);

  // 7) fused post-LN / l2norm / style-LN / scale-shift / SiLU -> bf16
  k_post_styl<<<ROWS_BT, 256, 0, stream>>>(attnf, g_post, b_post, g_st, b_st, embout, abf);

  // 8) style GEMM with fused residual: out = x + 0.1*(hs@W_out + b_out)
  k_w_to_bf16T<<<gWT, 256, 0, stream>>>(W_out, wbf, DD, DD);
  k_gemm_bf16<<<gGemm, 256, 0, stream>>>(abf, wbf, b_out, out, ROWS_BT, DD, DD, 1.0f, 2, x);
}
